// CausalSelfAttentionBatched_53609781788777
// MI455X (gfx1250) — matrix operand order repaired and checked by reading, not compiled, not run on hardware
//
#include <hip/hip_runtime.h>

// ---------------------------------------------------------------------------
// CDNA5 (gfx1250, wave32) causal self-attention block.
// All GEMM-like math runs on v_wmma_f32_16x16x32_bf16 (fp32 accumulate).
// ---------------------------------------------------------------------------

#define BATCH      4
#define SEQ        2048
#define DIMD       1024
#define NHEADS     8
#define HDIM       128
#define C3         3072          // 3 * NHEADS * HDIM
#define ATTN_SCALE 0.12f
#define RMS_EPS    1.1920929e-07f
#define NEGINF     (-1e30f)

typedef __bf16 bf16_t;
typedef __attribute__((ext_vector_type(16))) __bf16 v16bf;
typedef __attribute__((ext_vector_type(8)))  __bf16 v8bf;
typedef __attribute__((ext_vector_type(8)))  float  v8f;

__device__ __forceinline__ unsigned short f2bfbits(float f) {
    unsigned u = __float_as_uint(f);
    // round-to-nearest-even
    unsigned r = u + 0x7fffu + ((u >> 16) & 1u);
    return (unsigned short)(r >> 16);
}
__device__ __forceinline__ bf16_t f2bf(float f) {
    unsigned short s = f2bfbits(f);
    bf16_t h; __builtin_memcpy(&h, &s, 2); return h;
}
__device__ __forceinline__ bf16_t bits2bf(unsigned short s) {
    bf16_t h; __builtin_memcpy(&h, &s, 2); return h;
}
__device__ __forceinline__ float bf2f(bf16_t h) {
    unsigned short s; __builtin_memcpy(&s, &h, 2);
    return __uint_as_float(((unsigned)s) << 16);
}
__device__ __forceinline__ v8f vzero8() {
    v8f z;
    #pragma unroll
    for (int i = 0; i < 8; ++i) z[i] = 0.0f;
    return z;
}
__device__ __forceinline__ v16bf make_a_frag(v8bf lo, v8bf hi) {
    return __builtin_shufflevector(lo, hi, 0,1,2,3,4,5,6,7,8,9,10,11,12,13,14,15);
}

// ---------------------------------------------------------------------------
// Kernel 1: f32 -> bf16 conversion
// ---------------------------------------------------------------------------
__global__ void cvt_f32_to_bf16_kernel(const float* __restrict__ in,
                                       bf16_t* __restrict__ out, int n) {
    int i = blockIdx.x * blockDim.x + threadIdx.x;
    if (i < n) out[i] = f2bf(in[i]);
}

// ---------------------------------------------------------------------------
// Kernel 2: bf16 WMMA GEMM.  C[M,N] = A[M,K] * W[N,K]^T  (fp32 accumulate)
// Block = 256 threads = 8 waves; wave w owns a 32x64 tile
// (2 A-fragments x 4 B-fragments -> 8 WMMAs per K=32 step).
// A fragment layout (16-bit A 16x32): lane<16 -> K{0..15},
//                                     lane>=16 -> K{16..31}.
// B fragment layout (16-bit B 32x16): lane holds col n=(lane&15),
//                                     K = (lane>>4)*16 + e  -> contiguous.
// ---------------------------------------------------------------------------
template <bool OUT_BF16>
__global__ __launch_bounds__(256, 1)
void gemm_bf16_kernel(const bf16_t* __restrict__ A,
                      const bf16_t* __restrict__ W,
                      void* __restrict__ outp,
                      int M, int N, int K) {
    const int wave = threadIdx.x >> 5;
    const int lane = threadIdx.x & 31;
    const int nl = lane & 15;
    const int hb = lane >> 4;

    const int m0 = blockIdx.y * 256 + wave * 32;
    const int n0 = blockIdx.x * 64;

    v8f acc[2][4];
    #pragma unroll
    for (int i = 0; i < 2; ++i)
        #pragma unroll
        for (int j = 0; j < 4; ++j) acc[i][j] = vzero8();

    const bf16_t* arow0 = A + (size_t)(m0 + nl) * K + hb * 16;
    const bf16_t* arow1 = arow0 + (size_t)16 * K;
    const bf16_t* wb    = W + hb * 16;

    for (int kb = 0; kb < K; kb += 32) {
        v16bf a0 = make_a_frag(*(const v8bf*)(arow0 + kb),
                               *(const v8bf*)(arow0 + kb + 8));
        v16bf a1 = make_a_frag(*(const v8bf*)(arow1 + kb),
                               *(const v8bf*)(arow1 + kb + 8));
        #pragma unroll
        for (int j = 0; j < 4; ++j) {
            const bf16_t* wp = wb + (size_t)(n0 + j * 16 + nl) * K + kb;
            v16bf bfrag = *(const v16bf*)wp;
            acc[0][j] = __builtin_amdgcn_wmma_f32_16x16x32_bf16(
                false, a0, false, bfrag, (short)0, acc[0][j], false, false);
            acc[1][j] = __builtin_amdgcn_wmma_f32_16x16x32_bf16(
                false, a1, false, bfrag, (short)0, acc[1][j], false, false);
        }
    }

    #pragma unroll
    for (int i = 0; i < 2; ++i) {
        #pragma unroll
        for (int j = 0; j < 4; ++j) {
            #pragma unroll
            for (int v = 0; v < 8; ++v) {
                int row = m0 + i * 16 + v + hb * 8;
                int col = n0 + j * 16 + nl;
                if constexpr (OUT_BF16)
                    ((bf16_t*)outp)[(size_t)row * N + col] = f2bf(acc[i][j][v]);
                else
                    ((float*)outp)[(size_t)row * N + col] = acc[i][j][v];
            }
        }
    }
}

// ---------------------------------------------------------------------------
// Kernel 3: per (token, head): RMS-norm + RoPE on q,k (in place, bf16);
//           v = l0*v + l1*ve, stored transposed VT[(b,h),d][t] (bf16).
// One wave handles one (bt, h); lane owns elements {l, l+32, l+64, l+96}.
// RoPE pairs (i, i+64): pair (l, l+64) uses freq j=l (<32, active),
// pair (l+32, l+96) uses j=l+32 (>=32 -> cos=1, sin=0 -> identity).
// ---------------------------------------------------------------------------
__global__ __launch_bounds__(256)
void norm_rope_mix_kernel(bf16_t* __restrict__ qkv,
                          const float* __restrict__ ve,
                          const float* __restrict__ lam,
                          bf16_t* __restrict__ vt) {
    const int wave = threadIdx.x >> 5;
    const int lane = threadIdx.x & 31;
    int g  = blockIdx.x * (blockDim.x >> 5) + wave;   // 0 .. B*T*H-1
    int h  = g & (NHEADS - 1);
    int bt = g >> 3;
    int t  = bt & (SEQ - 1);
    int b  = bt >> 11;

    float theta = (float)t * exp2f(-10.0f * (float)lane * (1.0f / 31.0f));
    float co = cosf(theta), si = sinf(theta);

    #pragma unroll
    for (int qk = 0; qk < 2; ++qk) {
        bf16_t* base = qkv + (size_t)bt * C3 + qk * 1024 + h * HDIM;
        float a = bf2f(base[lane]);
        float e = bf2f(base[lane + 32]);
        float c = bf2f(base[lane + 64]);
        float d = bf2f(base[lane + 96]);
        float ss = a * a + e * e + c * c + d * d;
        #pragma unroll
        for (int m = 1; m < 32; m <<= 1) ss += __shfl_xor(ss, m, 32);
        float r = rsqrtf(ss * (1.0f / 128.0f) + RMS_EPS);
        a *= r; e *= r; c *= r; d *= r;
        float na =  a * co + c * si;
        float nc = -a * si + c * co;
        base[lane]      = f2bf(na);
        base[lane + 32] = f2bf(e);
        base[lane + 64] = f2bf(nc);
        base[lane + 96] = f2bf(d);
    }

    float l0 = lam[0], l1 = lam[1];
    const bf16_t* vbase  = qkv + (size_t)bt * C3 + 2048 + h * HDIM;
    const float*  vebase = ve + (size_t)bt * DIMD + h * HDIM;
    bf16_t* vtb = vt + ((size_t)(b * NHEADS + h) * HDIM) * SEQ + t;
    #pragma unroll
    for (int j = 0; j < 4; ++j) {
        int d = lane + j * 32;
        float vm = l0 * bf2f(vbase[d]) + l1 * vebase[d];
        vtb[(size_t)d * SEQ] = f2bf(vm);
    }
}

// ---------------------------------------------------------------------------
// Kernel 4: flash attention, one wave per (b, h, 16-query tile).
// Scores: S = Q K^T * 0.12 via 4 WMMAs per 16-key group (K-dim = head dim).
// Online softmax in the C-fragment layout; P transposed to A-layout
// through wave-private LDS; P*V via WMMA with VT (keys contiguous).
// launch_bounds(128,1): keep Q frags + 8 accum frags register-resident.
// ---------------------------------------------------------------------------
__global__ __launch_bounds__(128, 1)
void attention_kernel(const bf16_t* __restrict__ qkv,
                      const bf16_t* __restrict__ vt,
                      bf16_t* __restrict__ y) {
    __shared__ unsigned short plds[4][16 * 32];   // per-wave P staging (1 KB)

    const int wave = threadIdx.x >> 5;
    const int lane = threadIdx.x & 31;
    const int nl = lane & 15;
    const int hb = lane >> 4;

    int g  = blockIdx.x * 4 + wave;               // 0 .. B*H*(T/16)-1
    int qt = g & 127;
    int bh = g >> 7;
    int h  = bh & (NHEADS - 1);
    int b  = bh >> 3;
    int q0 = qt * 16;

    // Resident Q A-fragments (head dim split into 4 K=32 chunks)
    v16bf qa[4];
    {
        const bf16_t* qrow =
            qkv + (size_t)(b * SEQ + q0 + nl) * C3 + h * HDIM + hb * 16;
        #pragma unroll
        for (int db = 0; db < 4; ++db) {
            v8bf lo = *(const v8bf*)(qrow + db * 32);
            v8bf hi = *(const v8bf*)(qrow + db * 32 + 8);
            qa[db] = make_a_frag(lo, hi);
        }
    }

    v8f o[8];
    float mrow[8], lrow[8];
    #pragma unroll
    for (int f = 0; f < 8; ++f) o[f] = vzero8();
    #pragma unroll
    for (int v = 0; v < 8; ++v) { mrow[v] = NEGINF; lrow[v] = 0.0f; }

    unsigned short* pw = plds[wave];
    const bf16_t* kbase = qkv + (size_t)(b * SEQ) * C3 + 1024 + h * HDIM + hb * 16;
    const bf16_t* vtb   = vt + ((size_t)(b * NHEADS + h) * HDIM) * SEQ;

    const int kend = q0 + 16;
    for (int kt = 0; kt < kend; kt += 32) {
        // --- scores: two 16x16 fragments covering 32 keys ---
        v8f s[2];
        #pragma unroll
        for (int kg = 0; kg < 2; ++kg) {
            v8f acc = vzero8();
            const bf16_t* kp = kbase + (size_t)(kt + kg * 16 + nl) * C3;
            #pragma unroll
            for (int db = 0; db < 4; ++db) {
                v16bf kfrag = *(const v16bf*)(kp + db * 32);
                acc = __builtin_amdgcn_wmma_f32_16x16x32_bf16(
                    false, qa[db], false, kfrag, (short)0, acc, false, false);
            }
            s[kg] = acc;
        }

        // --- online softmax, per output row v (row = q0 + v + 8*hb) ---
        float sc[8];
        #pragma unroll
        for (int v = 0; v < 8; ++v) {
            int q = q0 + v + hb * 8;
            float s0 = s[0][v] * ATTN_SCALE; if (kt + nl      > q) s0 = NEGINF;
            float s1 = s[1][v] * ATTN_SCALE; if (kt + 16 + nl > q) s1 = NEGINF;
            float mt = fmaxf(s0, s1);
            #pragma unroll
            for (int m = 1; m < 16; m <<= 1) mt = fmaxf(mt, __shfl_xor(mt, m, 32));
            float mn = fmaxf(mrow[v], mt);
            sc[v] = __expf(mrow[v] - mn);
            float p0 = __expf(s0 - mn);
            float p1 = __expf(s1 - mn);
            float ps = p0 + p1;
            #pragma unroll
            for (int m = 1; m < 16; m <<= 1) ps += __shfl_xor(ps, m, 32);
            lrow[v] = lrow[v] * sc[v] + ps;
            mrow[v] = mn;
            int r = v + hb * 8;
            pw[r * 32 + nl]      = f2bfbits(p0);
            pw[r * 32 + 16 + nl] = f2bfbits(p1);
        }
        #pragma unroll
        for (int f = 0; f < 8; ++f)
            #pragma unroll
            for (int v = 0; v < 8; ++v) o[f][v] *= sc[v];

        // LDS writes are wave-private; order write->read within the wave.
        asm volatile("s_wait_dscnt 0" ::: "memory");

        // --- transpose P (C layout) -> A layout (16x32, K = key) ---
        v16bf pa;
        {
            const unsigned short* pr = pw + nl * 32 + hb * 16;
            #pragma unroll
            for (int e = 0; e < 8; ++e) pa[e]     = bits2bf(pr[e]);
            #pragma unroll
            for (int e = 0; e < 8; ++e) pa[8 + e] = bits2bf(pr[8 + e]);
        }

        // --- O += P * V  (8 fragments cover head dim 128) ---
        #pragma unroll
        for (int f = 0; f < 8; ++f) {
            const bf16_t* vp = vtb + (size_t)(f * 16 + nl) * SEQ + kt + hb * 16;
            v16bf vfrag = *(const v16bf*)vp;
            o[f] = __builtin_amdgcn_wmma_f32_16x16x32_bf16(
                false, pa, false, vfrag, (short)0, o[f], false, false);
        }
    }

    // --- finalize: divide by l, store bf16 into y[bt][h*128 + d] ---
    #pragma unroll
    for (int f = 0; f < 8; ++f) {
        #pragma unroll
        for (int v = 0; v < 8; ++v) {
            int q = q0 + v + hb * 8;
            y[(size_t)(b * SEQ + q) * DIMD + h * HDIM + f * 16 + nl] =
                f2bf(o[f][v] / lrow[v]);
        }
    }
}

// ---------------------------------------------------------------------------
// Host-side orchestration
// ---------------------------------------------------------------------------
extern "C" void kernel_launch(void* const* d_in, const int* in_sizes, int n_in,
                              void* d_out, int out_size, void* d_ws, size_t ws_size,
                              hipStream_t stream) {
    (void)in_sizes; (void)n_in; (void)out_size; (void)ws_size;

    const float* x     = (const float*)d_in[0];   // (4,2048,1024)
    const float* ve    = (const float*)d_in[1];   // (4,2048,1024)
    const float* lam   = (const float*)d_in[2];   // (2,)
    const float* qkv_w = (const float*)d_in[3];   // (3,1024,1024)
    const float* prj_w = (const float*)d_in[4];   // (1024,1024)
    float* out = (float*)d_out;

    char* w = (char*)d_ws;
    bf16_t* wqkv  = (bf16_t*)(w);                  //  6,291,456 B
    bf16_t* wproj = (bf16_t*)(w +  6291456);       //  2,097,152 B
    bf16_t* xb    = (bf16_t*)(w +  8388608);       // 16,777,216 B
    bf16_t* qkvb  = (bf16_t*)(w + 25165824);       // 50,331,648 B
    bf16_t* vtb   = (bf16_t*)(w + 75497472);       // 16,777,216 B
    bf16_t* yb    = (bf16_t*)(w + 92274688);       // 16,777,216 B

    const int n_x  = BATCH * SEQ * DIMD;           // 8,388,608
    const int n_wq = 3 * DIMD * DIMD;              // 3,145,728
    const int n_wp = DIMD * DIMD;                  // 1,048,576

    cvt_f32_to_bf16_kernel<<<(n_x  + 255) / 256, 256, 0, stream>>>(x,     xb,    n_x);
    cvt_f32_to_bf16_kernel<<<(n_wq + 255) / 256, 256, 0, stream>>>(qkv_w, wqkv,  n_wq);
    cvt_f32_to_bf16_kernel<<<(n_wp + 255) / 256, 256, 0, stream>>>(prj_w, wproj, n_wp);

    // QKV projection: (8192 x 3072) = x (8192 x 1024) * Wqkv^T
    gemm_bf16_kernel<true><<<dim3(C3 / 64, (BATCH * SEQ) / 256), 256, 0, stream>>>(
        xb, wqkv, qkvb, BATCH * SEQ, C3, DIMD);

    // RMS-norm + RoPE (q,k) + lambda-mix of v into transposed VT
    norm_rope_mix_kernel<<<(BATCH * SEQ * NHEADS) / 8, 256, 0, stream>>>(
        qkvb, ve, lam, vtb);

    // Flash attention: B*H*(T/16) = 4096 waves, 4 waves/block
    attention_kernel<<<(BATCH * NHEADS * (SEQ / 16)) / 4, 128, 0, stream>>>(
        qkvb, vtb, yb);

    // Output projection: (8192 x 1024) = y (8192 x 1024) * Wproj^T -> fp32
    gemm_bf16_kernel<false><<<dim3(DIMD / 64, (BATCH * SEQ) / 256), 256, 0, stream>>>(
        yb, wproj, out, BATCH * SEQ, DIMD, DIMD);
}
